// FUB_29308856828314
// MI455X (gfx1250) — matrix operand order repaired and checked by reading, not compiled, not run on hardware
//
#include <hip/hip_runtime.h>

typedef __attribute__((ext_vector_type(16))) __bf16 v16bf;
typedef __attribute__((ext_vector_type(8)))  __bf16 v8bf;
typedef __attribute__((ext_vector_type(8)))  float  v8f;
typedef __attribute__((ext_vector_type(4)))  float  v4f;

#define N_NODES 5
#define NB      4
#define NC      256
#define NHW     1024
#define KP      22          // f32 buffer: padded hw-row (16 data + 6 pad), bank-conflict-free
#define KPB     264         // bf16 buffer: padded k-row (256 data + 8 pad), 528 B rows (16B aligned)
#define THRESH  0.3f
#define EPSV    1e-12f

#define F32_BYTES  (N_NODES * NC * KP * 4)        // 112640
#define BF16_BYTES (N_NODES * 16 * KPB * 2)       // 42240
#define LDS_BYTES  (F32_BYTES + BF16_BYTES)       // 154880 < 320 KB

// Block = (i, b, hw-tile); 16 waves, wave w owns c-tile w.
// Phase 1: async-to-LDS copy of x[0..4, b, :, hw-tile] (f32, ASYNCcnt).
// Phase 2: one-time f32->bf16 repack into WMMA-B fragment order [j][hw][k].
// Phase 3: 8 K-chunks of v_wmma_f32_16x16x32_bf16 (5x A[i]@x[j] + W2[i]@x[i]);
//          B-frag = 2x ds_load_b128, zero per-wave conversion.
// Phase 4: fused sigmoid/threshold/L2-normalize/weighted-sum epilogue in fp32.
__global__ __launch_bounds__(512) void fused_graph_kernel(
    const float* __restrict__ x,       // (5,4,256,1024)
    const float* __restrict__ w,       // (5,5)
    const float* __restrict__ conv_w,  // (5,256,512)
    const float* __restrict__ conv_b,  // (5,256)
    float* __restrict__ out)           // (5,4,256,1024)
{
  extern __shared__ char smem[];
  float*  xsf = (float*)smem;                    // [j][k][KP]   f32 tile
  __bf16* xbf = (__bf16*)(smem + F32_BYTES);     // [j][hw][KPB] bf16 tile, frag-ordered

  const int tid  = threadIdx.x;
  const int lane = tid & 31;
  const int ct   = __builtin_amdgcn_readfirstlane(tid >> 5);  // wave id = c-tile

  const int blk = blockIdx.x;          // 5*4*64 = 1280 blocks
  const int ht  = blk & 63;
  const int b   = (blk >> 6) & 3;
  const int i   = blk >> 8;

  const int half = lane >> 4;
  const int l15  = lane & 15;

  // ---- Phase 1: async stage x[j, b, 0..255, ht*16 .. +15] -> LDS f32 ----
  // 5*256 rows of 16 f32 = 10240 8-byte pairs; 512 threads x 20 iters.
  for (int it = 0; it < 20; ++it) {
    const int pidx = tid + it * 512;
    const int j    = pidx >> 11;
    const int rem  = pidx & 2047;
    const int k    = rem >> 3;
    const int pw   = rem & 7;
    const float* g = x + (((size_t)j * NB + b) * NC + k) * NHW + ht * 16 + 2 * pw;
    const float* l = xsf + ((j * NC + k) * KP + 2 * pw);
    const unsigned laddr = (unsigned)(size_t)l;   // LDS offset = low 32 bits of flat addr
    asm volatile("global_load_async_to_lds_b64 %0, %1, off"
                 :: "v"(laddr), "v"(g) : "memory");
  }
  asm volatile("s_wait_asynccnt 0x0" ::: "memory");
  __syncthreads();

  // ---- Phase 2: one-time repack f32 -> bf16 in B-fragment order ----
  // 5*16 rows x 128 k-pairs = 10240 pairs; 512 threads x 20 iters.
  for (int it = 0; it < 20; ++it) {
    const int pidx = tid + it * 512;
    const int j    = pidx >> 11;
    const int rem  = pidx & 2047;
    const int hw   = rem >> 7;          // 0..15
    const int k0   = (rem & 127) * 2;   // even k
    const float f0 = xsf[(j * NC + k0)     * KP + hw];
    const float f1 = xsf[(j * NC + k0 + 1) * KP + hw];
    __bf16* d = xbf + ((j * 16 + hw) * KPB + k0);
    d[0] = (__bf16)f0;
    d[1] = (__bf16)f1;
  }
  __syncthreads();

  // ---- Phase 3: GEMM ----
  v8f acc[N_NODES] = {};               // G[i,j] tiles (f32 accum)
  v8f accI = {};                       // Gi tile (term_i)

  const int c_rowA = ct * 16 + l15;    // A-frag row M = lane&15
  const float* cw_row = conv_w + ((size_t)i * NC + c_rowA) * (2 * NC);

  for (int kc = 0; kc < 8; ++kc) {
    // A (=W1+W2) and W2 fragments in bf16.
    // lane<16 holds K {0..15}; lane>=16 holds K {16..31}
    const int ka = kc * 32 + half * 16;             // float4-aligned (multiple of 16)
    const v4f w1a = *(const v4f*)(cw_row + ka);
    const v4f w1b = *(const v4f*)(cw_row + ka + 4);
    const v4f w1c = *(const v4f*)(cw_row + ka + 8);
    const v4f w1d = *(const v4f*)(cw_row + ka + 12);
    const v4f w2a = *(const v4f*)(cw_row + NC + ka);
    const v4f w2b = *(const v4f*)(cw_row + NC + ka + 4);
    const v4f w2c = *(const v4f*)(cw_row + NC + ka + 8);
    const v4f w2d = *(const v4f*)(cw_row + NC + ka + 12);
    v16bf afrag, w2frag;
#pragma unroll
    for (int e = 0; e < 4; ++e) {
      afrag[e]      = (__bf16)(w1a[e] + w2a[e]);
      afrag[e + 4]  = (__bf16)(w1b[e] + w2b[e]);
      afrag[e + 8]  = (__bf16)(w1c[e] + w2c[e]);
      afrag[e + 12] = (__bf16)(w1d[e] + w2d[e]);
      w2frag[e]      = (__bf16)w2a[e];
      w2frag[e + 4]  = (__bf16)w2b[e];
      w2frag[e + 8]  = (__bf16)w2c[e];
      w2frag[e + 12] = (__bf16)w2d[e];
    }

    // B fragments: 32 contiguous bytes per lane from prepacked bf16 LDS.
    const int kb = kc * 32 + half * 16;             // 32-byte aligned offset
#pragma unroll
    for (int j = 0; j < N_NODES; ++j) {
      const v8bf* p = (const v8bf*)(xbf + ((j * 16 + l15) * KPB + kb));
      const v8bf lo = p[0];
      const v8bf hi = p[1];
      const v16bf bfrag = __builtin_shufflevector(
          lo, hi, 0, 1, 2, 3, 4, 5, 6, 7, 8, 9, 10, 11, 12, 13, 14, 15);

      acc[j] = __builtin_amdgcn_wmma_f32_16x16x32_bf16(
          false, afrag, false, bfrag, (short)0, acc[j], false, false);
      if (j == i)   // uniform scalar branch; EXEC stays all-1s around WMMA
        accI = __builtin_amdgcn_wmma_f32_16x16x32_bf16(
            false, w2frag, false, bfrag, (short)0, accI, false, false);
    }
  }

  // ---- Phase 4: epilogue. C/D layout: reg r holds M = r + 8*half, N = lane&15 ----
  float wrow[N_NODES];
#pragma unroll
  for (int j = 0; j < N_NODES; ++j) wrow[j] = w[i * N_NODES + j];

  const int hw = ht * 16 + l15;
#pragma unroll
  for (int r = 0; r < 8; ++r) {
    const int c      = ct * 16 + r + half * 8;
    const float bias = conv_b[i * NC + c];
    const float ti   = accI[r];

    float ssum = 0.f, num = 0.f;
#pragma unroll
    for (int j = 0; j < N_NODES; ++j) {
      const float xj     = xsf[(j * NC + c) * KP + l15];  // full-precision node value
      const float target = acc[j][r] + ti + bias;
      const float dist   = xj - target;
      const float z      = dist * wrow[j];
      float ev = 1.0f / (1.0f + __expf(-z));              // sigmoid
      ev = (ev > THRESH) ? ev : 0.0f;                     // threshold
      ssum += ev * ev;
      num  += ev * xj;
    }
    const float nrm  = __builtin_sqrtf(ssum);
    const float outv = num / fmaxf(nrm, EPSV);
    out[(((size_t)i * NB + b) * NC + c) * NHW + hw] = outv;
  }
}

extern "C" void kernel_launch(void* const* d_in, const int* in_sizes, int n_in,
                              void* d_out, int out_size, void* d_ws, size_t ws_size,
                              hipStream_t stream) {
  const float* x      = (const float*)d_in[0];
  const float* w      = (const float*)d_in[1];
  const float* conv_w = (const float*)d_in[2];
  const float* conv_b = (const float*)d_in[3];
  float* out = (float*)d_out;

  // 5 i * 4 b * 64 hw-tiles = 1280 blocks; 16 waves/block (one c-tile each)
  dim3 grid(1280), block(512);
  hipLaunchKernelGGL(fused_graph_kernel, grid, block, LDS_BYTES, stream,
                     x, w, conv_w, conv_b, out);
}
